// EncoderLayer_52192442581911
// MI455X (gfx1250) — compile-verified, matrix operand order checked by reading, not run on hardware
//
#include <hip/hip_runtime.h>
#include <hip/hip_bf16.h>

typedef unsigned short u16;
typedef __attribute__((ext_vector_type(16))) __bf16 v16bf;
typedef __attribute__((ext_vector_type(8)))  float  v8f;

__device__ __forceinline__ float bf2f(u16 u) {
  return __uint_as_float(((unsigned)u) << 16);
}
__device__ __forceinline__ u16 f2bf(float f) {
  unsigned u = __float_as_uint(f);
  u += 0x7FFFu + ((u >> 16) & 1u);   // round-to-nearest-even
  return (u16)(u >> 16);
}

// ---------------------------------------------------------------------------
// Generic f32 -> bf16 convert
// ---------------------------------------------------------------------------
__global__ void cvt_kernel(const float* __restrict__ src, u16* __restrict__ dst, int n) {
  int i = blockIdx.x * 256 + threadIdx.x;
  if (i < n) dst[i] = f2bf(src[i]);
}

// ---------------------------------------------------------------------------
// LayerNorm over 1024 cols, one block per row, writes bf16
// ---------------------------------------------------------------------------
__global__ void __launch_bounds__(256) ln_kernel(const float* __restrict__ in,
                                                 const float* __restrict__ g,
                                                 const float* __restrict__ b,
                                                 u16* __restrict__ out) {
  __shared__ float s1[256], s2[256];
  const int tid = threadIdx.x;
  const size_t row = blockIdx.x;
  const float* rp = in + row * 1024;
  float v[4], s = 0.f, ss = 0.f;
#pragma unroll
  for (int i = 0; i < 4; ++i) {
    v[i] = rp[tid + i * 256];
    s += v[i];
    ss += v[i] * v[i];
  }
  s1[tid] = s; s2[tid] = ss;
  __syncthreads();
  for (int o = 128; o > 0; o >>= 1) {
    if (tid < o) { s1[tid] += s1[tid + o]; s2[tid] += s2[tid + o]; }
    __syncthreads();
  }
  const float mean = s1[0] * (1.f / 1024.f);
  const float var  = s2[0] * (1.f / 1024.f) - mean * mean;
  const float rstd = rsqrtf(var + 1e-6f);
#pragma unroll
  for (int i = 0; i < 4; ++i) {
    int c = tid + i * 256;
    out[row * 1024 + c] = f2bf((v[i] - mean) * rstd * g[c] + b[c]);
  }
}

// ---------------------------------------------------------------------------
// bf16 WMMA GEMM: C[MxN] = A[MxK] @ B[KxN] (+bias)(+relu)(+residual)
// Tile 128x128x32, 256 threads = 8 waves, wave tile 64x32 (4x2 WMMA accs)
// A tile staged via CDNA5 async global->LDS DMA (ASYNCcnt); B tile staged
// with a 4x4 register-blocked transpose (b64 loads / b64 LDS stores).
// ---------------------------------------------------------------------------
#define BMT 128
#define BNT 128
#define BKT 32
#define TPAD 40   // padded LDS row stride (halfwords); 80B rows keep 16B align

__global__ void __launch_bounds__(256) gemm_bf16_kernel(
    const u16* __restrict__ A, const u16* __restrict__ B,
    const float* __restrict__ bias,
    float* __restrict__ Cf, u16* __restrict__ Cb,
    const float* __restrict__ residual,
    int M, int N, int K, int relu)
{
  __shared__ u16 As[BMT * TPAD];
  __shared__ u16 Bt[BNT * TPAD];

  const int tid  = threadIdx.x;
  const int lane = tid & 31;
  const int wave = tid >> 5;
  const int wm   = (wave >> 2) * 64;   // wave M offset within tile
  const int wn   = (wave & 3) * 32;    // wave N offset within tile
  const int half = lane >> 4;          // lane group (0: lanes0-15, 1: 16-31)
  const int l15  = lane & 15;

  const int mBase = blockIdx.y * BMT;
  const int nBase = blockIdx.x * BNT;

  // LDS byte address of As base (flat shared ptr low 32 bits == LDS offset)
  const unsigned asBase = (unsigned)(size_t)(&As[0]);

  // A staging geometry (4 chunks of 4 halfwords per thread), row clamped:
  // rows >= M only feed C rows >= M which are never stored.
  int aRow[4], aKo[4];
#pragma unroll
  for (int p = 0; p < 4; ++p) {
    int c = tid + p * 256;
    aRow[p] = c >> 3;
    aKo[p]  = (c & 7) * 4;
  }

  // B staging geometry: 4x4 block transpose
  const int bN0 = (tid & 31) * 4;
  const int bK0 = (tid >> 5) * 4;

  v8f acc[4][2];
#pragma unroll
  for (int i = 0; i < 4; ++i)
#pragma unroll
    for (int j = 0; j < 2; ++j)
#pragma unroll
      for (int r = 0; r < 8; ++r) acc[i][j][r] = 0.0f;

  for (int kc = 0; kc < K; kc += BKT) {
    if (kc + BKT < K) {  // prefetch next K-tile (global_prefetch_b8)
      __builtin_prefetch(A + (size_t)(mBase + (tid >> 1)) * K + kc + BKT + (tid & 1) * 16, 0, 0);
      __builtin_prefetch(B + (size_t)(kc + BKT + (tid >> 4)) * N + nBase + (tid & 15) * 8, 0, 0);
    }
    // ---- A tile: async DMA global -> LDS (no VGPR transit) ----
#pragma unroll
    for (int p = 0; p < 4; ++p) {
      int gr = mBase + aRow[p];
      if (gr > M - 1) gr = M - 1;
      unsigned long long ga =
          (unsigned long long)(size_t)(A + (size_t)gr * K + kc + aKo[p]);
      unsigned la = asBase + (unsigned)((aRow[p] * TPAD + aKo[p]) * 2);
      asm volatile("global_load_async_to_lds_b64 %0, %1, off"
                   :: "v"(la), "v"(ga) : "memory");
    }
    // ---- B tile: 4x4 register transpose, vectorized b64 both ways ----
    {
      union { unsigned long long u; u16 s[4]; } gv[4], ov[4];
#pragma unroll
      for (int i = 0; i < 4; ++i)
        gv[i].u = *(const unsigned long long*)(B + (size_t)(kc + bK0 + i) * N + nBase + bN0);
#pragma unroll
      for (int j = 0; j < 4; ++j) {
#pragma unroll
        for (int i = 0; i < 4; ++i) ov[j].s[i] = gv[i].s[j];
        *(unsigned long long*)(&Bt[(bN0 + j) * TPAD + bK0]) = ov[j].u;
      }
    }
    asm volatile("s_wait_asynccnt 0x0" ::: "memory");
    __syncthreads();

    // A fragment (16x32 bf16): lane holds row m=l15, k runs {half*8..+7, 16+half*8..+7}
    v16bf afr[4], bfr[2];
#pragma unroll
    for (int mt = 0; mt < 4; ++mt) {
      const u16* p = &As[(wm + mt * 16 + l15) * TPAD + half * 8];
      union { v16bf v; uint4 q[2]; } u;
      u.q[0] = *(const uint4*)(p);
      u.q[1] = *(const uint4*)(p + 16);
      afr[mt] = u.v;
    }
#pragma unroll
    for (int nt = 0; nt < 2; ++nt) {
      const u16* p = &Bt[(wn + nt * 16 + l15) * TPAD + half * 8];
      union { v16bf v; uint4 q[2]; } u;
      u.q[0] = *(const uint4*)(p);
      u.q[1] = *(const uint4*)(p + 16);
      bfr[nt] = u.v;
    }
#pragma unroll
    for (int mt = 0; mt < 4; ++mt)
#pragma unroll
      for (int nt = 0; nt < 2; ++nt)
        acc[mt][nt] = __builtin_amdgcn_wmma_f32_16x16x32_bf16(
            false, afr[mt], false, bfr[nt], (short)0, acc[mt][nt], false, false);
    __syncthreads();
  }

  // epilogue: C/D layout -> row = half*8 + r, col = l15
#pragma unroll
  for (int mt = 0; mt < 4; ++mt) {
#pragma unroll
    for (int nt = 0; nt < 2; ++nt) {
      const int col = nBase + wn + nt * 16 + l15;
      const float bv = bias ? bias[col] : 0.f;
#pragma unroll
      for (int r = 0; r < 8; ++r) {
        const int row = mBase + wm + mt * 16 + half * 8 + r;
        if (row < M) {
          float val = acc[mt][nt][r] + bv;
          if (relu) val = fmaxf(val, 0.f);
          const size_t idx = (size_t)row * N + col;
          if (residual) val += residual[idx];
          if (Cf) Cf[idx] = val;
          if (Cb) Cb[idx] = f2bf(val);
        }
      }
    }
  }
}

// ---------------------------------------------------------------------------
// Stage-1 anchor scores: sc1[b,h,a,n] = scale * sum_d qa[h,a,d]*k[b,n,h,d]
// one thread per (b,h,n)
// ---------------------------------------------------------------------------
__global__ void __launch_bounds__(256) scores1_kernel(const u16* __restrict__ qkv,
                                                      const float* __restrict__ qa,
                                                      float* __restrict__ sc1) {
  int idx = blockIdx.x * 256 + threadIdx.x;
  if (idx >= 8 * 8 * 2048) return;
  const int n = idx & 2047;
  const int h = (idx >> 11) & 7;
  const int b = idx >> 14;
  const u16* kp = qkv + (((size_t)(b * 2048 + n) * 3 + 1) << 10) + h * 128;
  const float* qp = qa + (size_t)h * 8 * 128;
  float acc[8];
#pragma unroll
  for (int a = 0; a < 8; ++a) acc[a] = 0.f;
  for (int d = 0; d < 128; ++d) {
    const float kd = bf2f(kp[d]);
#pragma unroll
    for (int a = 0; a < 8; ++a) acc[a] += kd * qp[a * 128 + d];
  }
  const float scale = 0.08838834764831845f;  // 1/sqrt(128)
#pragma unroll
  for (int a = 0; a < 8; ++a)
    sc1[(((size_t)(b * 8 + h) * 8 + a) << 11) + n] = acc[a] * scale;
}

// ---------------------------------------------------------------------------
// In-place softmax over rows of length `len` (block per row)
// ---------------------------------------------------------------------------
__global__ void __launch_bounds__(256) softmax_rows_kernel(float* __restrict__ data, int len) {
  __shared__ float red[256];
  const int tid = threadIdx.x;
  float* row = data + (size_t)blockIdx.x * len;
  float m = -3.4e38f;
  for (int i = tid; i < len; i += 256) m = fmaxf(m, row[i]);
  red[tid] = m; __syncthreads();
  for (int o = 128; o > 0; o >>= 1) {
    if (tid < o) red[tid] = fmaxf(red[tid], red[tid + o]);
    __syncthreads();
  }
  m = red[0]; __syncthreads();
  float s = 0.f;
  for (int i = tid; i < len; i += 256) {
    float e = __expf(row[i] - m);
    row[i] = e;
    s += e;
  }
  red[tid] = s; __syncthreads();
  for (int o = 128; o > 0; o >>= 1) {
    if (tid < o) red[tid] += red[tid + o];
    __syncthreads();
  }
  const float inv = 1.f / red[0];
  for (int i = tid; i < len; i += 256) row[i] *= inv;
}

// ---------------------------------------------------------------------------
// out1[b,h,a*128+d] = sum_n sc1[b,h,a,n] * v[b,n,h,d]; block per (b,h,a)
// ---------------------------------------------------------------------------
__global__ void __launch_bounds__(128) out1_kernel(const float* __restrict__ sc1,
                                                   const u16* __restrict__ qkv,
                                                   u16* __restrict__ out1) {
  const int bha = blockIdx.x;       // 512 = b*64 + h*8 + a
  const int a = bha & 7;
  const int h = (bha >> 3) & 7;
  const int b = bha >> 6;
  const int d = threadIdx.x;        // 128
  const float* p = sc1 + ((size_t)bha << 11);
  float acc = 0.f;
  for (int n = 0; n < 2048; ++n) {
    const float w = p[n];
    const float v = bf2f(qkv[(((size_t)(b * 2048 + n) * 3 + 2) << 10) + h * 128 + d]);
    acc += w * v;
  }
  out1[((size_t)(b * 8 + h) << 10) + a * 128 + d] = f2bf(acc);
}

// ---------------------------------------------------------------------------
// (b,H,NA,512) -> (b,NA,H,512) bf16 transpose; thread per dst element
// ---------------------------------------------------------------------------
__global__ void tnet_kernel(const u16* __restrict__ han, u16* __restrict__ tnet) {
  int idx = blockIdx.x * 256 + threadIdx.x;  // 64*4096
  if (idx >= 262144) return;
  const int c = idx & 511;
  const int h = (idx >> 9) & 7;
  const int a = (idx >> 12) & 7;
  const int b = idx >> 15;
  tnet[idx] = han[(((size_t)(b * 8 + h)) << 12) + a * 512 + c];
}

// ---------------------------------------------------------------------------
// Fused stage-2 attention: per (b,h,n): 8 dots vs kk, softmax(8), mix ov -> y
// ---------------------------------------------------------------------------
__global__ void __launch_bounds__(256) attn2_kernel(const u16* __restrict__ qkv,
                                                    const float* __restrict__ kk,
                                                    const float* __restrict__ ov,
                                                    u16* __restrict__ y) {
  int idx = blockIdx.x * 256 + threadIdx.x;
  if (idx >= 8 * 8 * 2048) return;
  const int n = idx & 2047;
  const int h = (idx >> 11) & 7;
  const int b = idx >> 14;
  const u16* qp = qkv + (((size_t)(b * 2048 + n) * 3) << 10) + h * 128;
  float acc[8];
#pragma unroll
  for (int a = 0; a < 8; ++a) acc[a] = 0.f;
  for (int d = 0; d < 128; ++d) {
    const float qd = bf2f(qp[d]);
#pragma unroll
    for (int a = 0; a < 8; ++a)
      acc[a] += qd * kk[((size_t)(b * 8 + a) << 10) + h * 128 + d];
  }
  const float scale = 0.08838834764831845f;
  float m = -3.4e38f;
#pragma unroll
  for (int a = 0; a < 8; ++a) { acc[a] *= scale; m = fmaxf(m, acc[a]); }
  float s = 0.f;
#pragma unroll
  for (int a = 0; a < 8; ++a) { acc[a] = __expf(acc[a] - m); s += acc[a]; }
  const float inv = 1.f / s;
#pragma unroll
  for (int a = 0; a < 8; ++a) acc[a] *= inv;
  u16* yp = y + ((size_t)(b * 2048 + n) << 10) + h * 128;
  for (int d = 0; d < 128; ++d) {
    float v = 0.f;
#pragma unroll
    for (int a = 0; a < 8; ++a)
      v += acc[a] * ov[((size_t)(b * 8 + a) << 10) + h * 128 + d];
    yp[d] = f2bf(v);
  }
}

// ---------------------------------------------------------------------------
// Host orchestration
// ---------------------------------------------------------------------------
extern "C" void kernel_launch(void* const* d_in, const int* in_sizes, int n_in,
                              void* d_out, int out_size, void* d_ws, size_t ws_size,
                              hipStream_t stream) {
  (void)in_sizes; (void)n_in; (void)out_size; (void)ws_size;
  const float* inputs   = (const float*)d_in[0];
  const float* ln_g     = (const float*)d_in[1];
  const float* ln_b     = (const float*)d_in[2];
  const float* q_anchor = (const float*)d_in[3];
  const float* W_qkv    = (const float*)d_in[4];
  const float* b_qkv    = (const float*)d_in[5];
  const float* W_an     = (const float*)d_in[6];
  const float* b_an     = (const float*)d_in[7];
  const float* W_n1     = (const float*)d_in[8];
  const float* b_n1     = (const float*)d_in[9];
  const float* W_n2     = (const float*)d_in[10];
  const float* b_n2     = (const float*)d_in[11];
  const float* W_n3     = (const float*)d_in[12];
  const float* b_n3     = (const float*)d_in[13];
  const float* W_k      = (const float*)d_in[14];
  const float* b_k      = (const float*)d_in[15];
  const float* W_o      = (const float*)d_in[16];
  const float* b_o      = (const float*)d_in[17];
  float* out = (float*)d_out;

  char* ws = (char*)d_ws;
  size_t off = 0;
  auto alloc = [&](size_t bytes) -> void* {
    void* p = ws + off;
    off += (bytes + 255) & ~(size_t)255;
    return p;
  };

  const int M = 8 * 2048;  // 16384 token rows
  u16* x_bf    = (u16*)alloc((size_t)M * 1024 * 2);
  u16* qkv_bf  = (u16*)alloc((size_t)M * 3072 * 2);
  u16* y_bf    = (u16*)alloc((size_t)M * 1024 * 2);
  u16* Wqkv_bf = (u16*)alloc((size_t)1024 * 3072 * 2);
  u16* Wan_bf  = (u16*)alloc((size_t)1024 * 4096 * 2);
  u16* Wn1_bf  = (u16*)alloc((size_t)4096 * 1024 * 2);
  u16* Wn2_bf  = (u16*)alloc((size_t)1024 * 4096 * 2);
  u16* Wn3_bf  = (u16*)alloc((size_t)4096 * 1024 * 2);
  u16* Wk_bf   = (u16*)alloc((size_t)1024 * 1024 * 2);
  u16* Wo_bf   = (u16*)alloc((size_t)1024 * 1024 * 2);
  float* sc1   = (float*)alloc((size_t)512 * 2048 * 4);
  u16* out1_bf = (u16*)alloc((size_t)64 * 1024 * 2);
  u16* han_bf  = (u16*)alloc((size_t)64 * 4096 * 2);
  u16* tnet_bf = (u16*)alloc((size_t)64 * 4096 * 2);
  u16* h1_bf   = (u16*)alloc((size_t)64 * 1024 * 2);
  u16* h2_bf   = (u16*)alloc((size_t)64 * 4096 * 2);
  u16* net_bf  = (u16*)alloc((size_t)64 * 1024 * 2);
  float* net_f = (float*)alloc((size_t)64 * 1024 * 4);
  float* kk_f  = (float*)alloc((size_t)64 * 1024 * 4);

  auto cvt = [&](const float* s, u16* d, int n) {
    cvt_kernel<<<(n + 255) / 256, 256, 0, stream>>>(s, d, n);
  };
  cvt(W_qkv, Wqkv_bf, 1024 * 3072);
  cvt(W_an,  Wan_bf,  1024 * 4096);
  cvt(W_n1,  Wn1_bf,  4096 * 1024);
  cvt(W_n2,  Wn2_bf,  1024 * 4096);
  cvt(W_n3,  Wn3_bf,  4096 * 1024);
  cvt(W_k,   Wk_bf,   1024 * 1024);
  cvt(W_o,   Wo_bf,   1024 * 1024);

  // 1) LayerNorm -> bf16 x
  ln_kernel<<<M, 256, 0, stream>>>(inputs, ln_g, ln_b, x_bf);

  // 2) qkv = x @ W_qkv + b_qkv  (dominant GEMM, bf16 WMMA)
  gemm_bf16_kernel<<<dim3(3072 / 128, M / 128), 256, 0, stream>>>(
      x_bf, Wqkv_bf, b_qkv, nullptr, qkv_bf, nullptr, M, 3072, 1024, 0);

  // 3) anchor attention
  scores1_kernel<<<131072 / 256, 256, 0, stream>>>(qkv_bf, q_anchor, sc1);
  softmax_rows_kernel<<<512, 256, 0, stream>>>(sc1, 2048);
  out1_kernel<<<512, 128, 0, stream>>>(sc1, qkv_bf, out1_bf);

  // 4) MLP stack (M=64 rows, masked tiles)
  gemm_bf16_kernel<<<dim3(4096 / 128, 1), 256, 0, stream>>>(
      out1_bf, Wan_bf, b_an, nullptr, han_bf, nullptr, 64, 4096, 1024, 1);
  tnet_kernel<<<262144 / 256, 256, 0, stream>>>(han_bf, tnet_bf);
  gemm_bf16_kernel<<<dim3(1024 / 128, 1), 256, 0, stream>>>(
      tnet_bf, Wn1_bf, b_n1, nullptr, h1_bf, nullptr, 64, 1024, 4096, 0);
  gemm_bf16_kernel<<<dim3(4096 / 128, 1), 256, 0, stream>>>(
      h1_bf, Wn2_bf, b_n2, nullptr, h2_bf, nullptr, 64, 4096, 1024, 1);
  gemm_bf16_kernel<<<dim3(1024 / 128, 1), 256, 0, stream>>>(
      h2_bf, Wn3_bf, b_n3, net_f, net_bf, nullptr, 64, 1024, 4096, 0);
  gemm_bf16_kernel<<<dim3(1024 / 128, 1), 256, 0, stream>>>(
      net_bf, Wk_bf, b_k, kk_f, nullptr, nullptr, 64, 1024, 1024, 0);

  // 5) stage-2 attention (fused dots + softmax(8) + mix)
  attn2_kernel<<<131072 / 256, 256, 0, stream>>>(qkv_bf, kk_f, net_f, y_bf);

  // 6) final projection + bias + residual -> d_out (f32)
  gemm_bf16_kernel<<<dim3(1024 / 128, M / 128), 256, 0, stream>>>(
      y_bf, Wo_bf, b_o, out, nullptr, inputs, M, 1024, 1024, 0);
}
